// Block_24249385353653
// MI455X (gfx1250) — matrix operand order repaired and checked by reading, not compiled, not run on hardware
//
#include <hip/hip_runtime.h>
#include <math.h>

// ---------- gfx1250 async global->LDS support (guarded; falls back to plain loads) ----------
#ifndef HAVE_ASYNC_LDS
#if defined(__has_builtin)
#if __has_builtin(__builtin_amdgcn_global_load_async_to_lds_b128)
#define HAVE_ASYNC_LDS 1
#endif
#endif
#endif
#ifndef HAVE_ASYNC_LDS
#define HAVE_ASYNC_LDS 0
#endif

typedef __attribute__((ext_vector_type(4))) int v4i;
#if HAVE_ASYNC_LDS
typedef __attribute__((address_space(1))) v4i* gv4p;   // global (__device__) int4*
typedef __attribute__((address_space(3))) v4i* lv4p;   // LDS (__shared__) int4*
#endif

// ---------- types ----------
typedef __attribute__((ext_vector_type(16))) __bf16 bf16x16;
typedef __attribute__((ext_vector_type(8)))  float  floatx8;

struct alignas(16) U4 { unsigned int x, y, z, w; };
union Frag {
  U4 q[2];
  bf16x16 v;
  unsigned short u[16];
};

__device__ __forceinline__ unsigned short f2bf(float f) {
  unsigned int u = __float_as_uint(f);
  u += 0x7fffu + ((u >> 16) & 1u);          // round-to-nearest-even
  return (unsigned short)(u >> 16);
}

__device__ __forceinline__ floatx8 zero8() {
  floatx8 z = {0.f, 0.f, 0.f, 0.f, 0.f, 0.f, 0.f, 0.f};
  return z;
}

__device__ __forceinline__ float gelu_exact(float x) {
  return 0.5f * x * (1.0f + erff(x * 0.70710678118654752440f));
}

// ---------- weight convert: W[K,N] f32 -> Wt[N,K] bf16 (LDS-tiled transpose) ----------
__global__ void wconv_kernel(const float* __restrict__ W, unsigned short* __restrict__ Wt,
                             int K, int N) {
  __shared__ unsigned short tile[16][17];
  int n0 = blockIdx.x * 16, k0 = blockIdx.y * 16;
  int tx = threadIdx.x, ty = threadIdx.y;
  tile[ty][tx] = f2bf(W[(size_t)(k0 + ty) * N + (n0 + tx)]);
  __syncthreads();
  Wt[(size_t)(n0 + ty) * K + (k0 + tx)] = tile[tx][ty];
}

// ---------- layernorm: one row per block, write bf16 ----------
__global__ void ln_kernel(const float* __restrict__ x, const float* __restrict__ g,
                          const float* __restrict__ b, unsigned short* __restrict__ out, int C) {
  __shared__ float s1[256], s2[256];
  int row = blockIdx.x;
  const float* xr = x + (size_t)row * C;
  float a = 0.f, sq = 0.f;
  for (int c = threadIdx.x; c < C; c += 256) { float v = xr[c]; a += v; sq += v * v; }
  s1[threadIdx.x] = a; s2[threadIdx.x] = sq;
  __syncthreads();
  for (int st = 128; st > 0; st >>= 1) {
    if ((int)threadIdx.x < st) { s1[threadIdx.x] += s1[threadIdx.x + st];
                                 s2[threadIdx.x] += s2[threadIdx.x + st]; }
    __syncthreads();
  }
  float mean = s1[0] / (float)C;
  float var  = s2[0] / (float)C - mean * mean;
  float rstd = rsqrtf(var + 1e-5f);
  for (int c = threadIdx.x; c < C; c += 256) {
    float v = (xr[c] - mean) * rstd * g[c] + b[c];
    out[(size_t)row * C + c] = f2bf(v);
  }
}

// ---------- WMMA GEMM, ping-pong pipeline; K and MODE compile-time ----------
// A[M,K] (lda == K for all uses), Bt[N,K] bf16. out = act(A @ Bt^T + bias) (+res)
// MODE 0: store bf16 ; 1: GELU then bf16 ; 2: fp32 store of (acc+bias+res)
template <int K, int MODE>
__global__ void gemm_bf16_kernel(const unsigned short* __restrict__ A,
                                 const unsigned short* __restrict__ Bt,
                                 const float* __restrict__ bias,
                                 unsigned short* __restrict__ outB,
                                 float* __restrict__ outF,
                                 const float* __restrict__ res,
                                 int ldo) {
  const int wave = threadIdx.x >> 5;
  const int lane = threadIdx.x & 31;
  const int hi = lane >> 4, lo = lane & 15;
  const int m0 = blockIdx.y * 64 + wave * 16;
  const int n0 = blockIdx.x * 64;

  floatx8 acc0 = zero8(), acc1 = zero8(), acc2 = zero8(), acc3 = zero8();

  // only two live per-lane pointers; all tile/step offsets fold into load immediates
  const unsigned short* aptr = A  + (size_t)(m0 + lo) * K + hi * 16;
  const unsigned short* bp   = Bt + (size_t)(n0 + lo) * K + hi * 16;

  auto ldA = [](Frag& f, const unsigned short* p, int ofs) {
    f.q[0] = *(const U4*)(p + ofs);         // K = ofs+hi*16 .. +7
    f.q[1] = *(const U4*)(p + ofs + 8);     // K = ofs+hi*16+8 .. +15
  };
  auto ldB = [](Frag& f, const unsigned short* p, int ofs) {
    f.q[0] = *(const U4*)(p + ofs);         // K = ofs+hi*16 .. +7
    f.q[1] = *(const U4*)(p + ofs + 8);     // K = ofs+hi*16+8 .. +15
  };

  Frag a0, a1, b00, b01, b02, b03, b10, b11, b12, b13;
  ldA(a0, aptr, 0);
  ldB(b00, bp, 0 * 16 * K); ldB(b01, bp, 1 * 16 * K);
  ldB(b02, bp, 2 * 16 * K); ldB(b03, bp, 3 * 16 * K);

  // K is a multiple of 64: 2x unrolled ping-pong, no register rotation copies
  for (int k0 = 0; k0 < K; k0 += 64) {
    ldA(a1, aptr, 32);
    ldB(b10, bp, 0 * 16 * K + 32); ldB(b11, bp, 1 * 16 * K + 32);
    ldB(b12, bp, 2 * 16 * K + 32); ldB(b13, bp, 3 * 16 * K + 32);
    __builtin_prefetch(aptr + 64, 0, 1);    // global_prefetch_b8 one step ahead
    acc0 = __builtin_amdgcn_wmma_f32_16x16x32_bf16(false, a0.v, false, b00.v, (short)0, acc0, false, false);
    acc1 = __builtin_amdgcn_wmma_f32_16x16x32_bf16(false, a0.v, false, b01.v, (short)0, acc1, false, false);
    acc2 = __builtin_amdgcn_wmma_f32_16x16x32_bf16(false, a0.v, false, b02.v, (short)0, acc2, false, false);
    acc3 = __builtin_amdgcn_wmma_f32_16x16x32_bf16(false, a0.v, false, b03.v, (short)0, acc3, false, false);
    aptr += 64; bp += 64;
    if (k0 + 64 < K) {
      ldA(a0, aptr, 0);
      ldB(b00, bp, 0 * 16 * K); ldB(b01, bp, 1 * 16 * K);
      ldB(b02, bp, 2 * 16 * K); ldB(b03, bp, 3 * 16 * K);
    }
    acc0 = __builtin_amdgcn_wmma_f32_16x16x32_bf16(false, a1.v, false, b10.v, (short)0, acc0, false, false);
    acc1 = __builtin_amdgcn_wmma_f32_16x16x32_bf16(false, a1.v, false, b11.v, (short)0, acc1, false, false);
    acc2 = __builtin_amdgcn_wmma_f32_16x16x32_bf16(false, a1.v, false, b12.v, (short)0, acc2, false, false);
    acc3 = __builtin_amdgcn_wmma_f32_16x16x32_bf16(false, a1.v, false, b13.v, (short)0, acc3, false, false);
  }

  floatx8 accv[4] = {acc0, acc1, acc2, acc3};
#pragma unroll
  for (int nt = 0; nt < 4; ++nt) {
    int n = n0 + nt * 16 + lo;
    float bval = bias[n];
#pragma unroll
    for (int v = 0; v < 8; ++v) {
      int m = m0 + v + 8 * hi;
      float val = accv[nt][v] + bval;
      size_t idx = (size_t)m * ldo + n;
      if (MODE == 0) {
        outB[idx] = f2bf(val);
      } else if (MODE == 1) {
        outB[idx] = f2bf(gelu_exact(val));
      } else {
        outF[idx] = val + res[idx];
      }
    }
  }
}

// ---------- flash attention: one wave per (b, h, 16-query tile) ----------
// qkv bf16 [B,T,3C]; q at +0, k at +C, v at +2C within the 3C row. D=64, scale 1/8.
// V tiles (32 keys x 64 d) are staged into double-buffered LDS via async
// global->LDS (ASYNCcnt) so the key-major gather hits LDS, not VMEM.
#define VP 72   // padded LDS row stride (halfwords) -> conflict-free column reads
__global__ void attn_kernel(const unsigned short* __restrict__ qkv,
                            unsigned short* __restrict__ y,
                            int T, int C3, int C) {
  __shared__ unsigned short Plds[16 * 32];
  __shared__ unsigned short Vlds[2][32 * VP];
  const int lane = threadIdx.x;
  const int hi = lane >> 4, lo = lane & 15;
  const int qt = blockIdx.x, h = blockIdx.y, b = blockIdx.z;

  // Q fragments (A-layout) for d=0..31 and d=32..63
  Frag aq0, aq1;
  {
    const unsigned short* qrow = qkv + ((size_t)(b * T + qt * 16 + lo)) * C3 + h * 64;
    aq0.q[0] = *(const U4*)(qrow + hi * 16);
    aq0.q[1] = *(const U4*)(qrow + 8 + hi * 16);
    aq1.q[0] = *(const U4*)(qrow + 32 + hi * 16);
    aq1.q[1] = *(const U4*)(qrow + 40 + hi * 16);
  }

  // stage one 32x64 bf16 V tile into LDS buffer `buf` (coalesced 16B chunks)
  auto stageV = [&](int kb, int buf) {
    const unsigned short* vb = qkv + ((size_t)(b * T + kb)) * C3 + 2 * C + h * 64;
#pragma unroll
    for (int it = 0; it < 8; ++it) {
      int c = it * 32 + lane;        // 256 chunks of 16B
      int row = c >> 3;              // key row 0..31
      int col = (c & 7) * 8;         // d offset 0..56
      const unsigned short* src = vb + (size_t)row * C3 + col;
      unsigned short* dst = &Vlds[buf][row * VP + col];
#if HAVE_ASYNC_LDS
      __builtin_amdgcn_global_load_async_to_lds_b128((gv4p)src, (lv4p)dst, 0, 0);
#else
      *(U4*)dst = *(const U4*)src;
#endif
    }
  };

  float mrow[8], lrow[8];
  floatx8 o[4];
#pragma unroll
  for (int v = 0; v < 8; ++v) { mrow[v] = -3.0e38f; lrow[v] = 0.f; }
  o[0] = zero8(); o[1] = zero8(); o[2] = zero8(); o[3] = zero8();

  const int kend = qt * 16 + 16;              // exclusive causal key bound
  const int nblocks = (kend + 31) / 32;

  stageV(0, 0);
#if HAVE_ASYNC_LDS
  asm volatile("s_wait_asynccnt 0x0" ::: "memory");
#endif
  __syncthreads();

  for (int blk = 0; blk < nblocks; ++blk) {
    const int kb = blk * 32;
    const int cur = blk & 1;
    if (blk + 1 < nblocks) stageV(kb + 32, cur ^ 1);   // overlap next V tile with compute

    // ---- S = Q @ K^T for 32 keys: two 16x16 tiles ----
    // load all four K B-fragments first so the four WMMAs issue back-to-back
    floatx8 s0 = zero8(), s1 = zero8();
    {
      const unsigned short* kr0 = qkv + ((size_t)(b * T + kb + lo)) * C3 + C + h * 64;
      const unsigned short* kr1 = kr0 + (size_t)16 * C3;
      Frag bk00, bk01, bk10, bk11;
      bk00.q[0] = ((const U4*)(kr0 + hi * 16))[0];
      bk00.q[1] = ((const U4*)(kr0 + hi * 16))[1];
      bk01.q[0] = ((const U4*)(kr0 + 32 + hi * 16))[0];
      bk01.q[1] = ((const U4*)(kr0 + 32 + hi * 16))[1];
      bk10.q[0] = ((const U4*)(kr1 + hi * 16))[0];
      bk10.q[1] = ((const U4*)(kr1 + hi * 16))[1];
      bk11.q[0] = ((const U4*)(kr1 + 32 + hi * 16))[0];
      bk11.q[1] = ((const U4*)(kr1 + 32 + hi * 16))[1];
      if (blk + 1 < nblocks)
        __builtin_prefetch(kr0 + 32 * C3, 0, 1);       // next key-block K rows
      s0 = __builtin_amdgcn_wmma_f32_16x16x32_bf16(false, aq0.v, false, bk00.v, (short)0, s0, false, false);
      s0 = __builtin_amdgcn_wmma_f32_16x16x32_bf16(false, aq1.v, false, bk01.v, (short)0, s0, false, false);
      s1 = __builtin_amdgcn_wmma_f32_16x16x32_bf16(false, aq0.v, false, bk10.v, (short)0, s1, false, false);
      s1 = __builtin_amdgcn_wmma_f32_16x16x32_bf16(false, aq1.v, false, bk11.v, (short)0, s1, false, false);
    }

    // ---- online softmax (rows live at VGPR v = m - 8*hi, cols across 16 lanes) ----
    float alpha[8];
#pragma unroll
    for (int v = 0; v < 8; ++v) {
      int qi = qt * 16 + hi * 8 + v;
      float sv0 = s0[v] * 0.125f; if (kb + lo > qi)       sv0 = -3.0e38f;
      float sv1 = s1[v] * 0.125f; if (kb + 16 + lo > qi)  sv1 = -3.0e38f;
      float bm = fmaxf(sv0, sv1);
      bm = fmaxf(bm, __shfl_xor(bm, 1, 32));
      bm = fmaxf(bm, __shfl_xor(bm, 2, 32));
      bm = fmaxf(bm, __shfl_xor(bm, 4, 32));
      bm = fmaxf(bm, __shfl_xor(bm, 8, 32));
      float nm = fmaxf(mrow[v], bm);
      float al = __expf(mrow[v] - nm);
      mrow[v] = nm;
      float p0 = __expf(sv0 - nm);
      float p1 = __expf(sv1 - nm);
      float rs = p0 + p1;
      rs += __shfl_xor(rs, 1, 32);
      rs += __shfl_xor(rs, 2, 32);
      rs += __shfl_xor(rs, 4, 32);
      rs += __shfl_xor(rs, 8, 32);
      lrow[v] = lrow[v] * al + rs;
      alpha[v] = al;
      Plds[(hi * 8 + v) * 32 + lo]      = f2bf(p0);
      Plds[(hi * 8 + v) * 32 + 16 + lo] = f2bf(p1);
    }
#pragma unroll
    for (int nt = 0; nt < 4; ++nt)
#pragma unroll
      for (int v = 0; v < 8; ++v) o[nt][v] *= alpha[v];

    __syncthreads();
    // re-stage P as A-fragment (16 rows x 32 keys)
    Frag pa;
    {
      const unsigned short* pr = &Plds[lo * 32];
      pa.q[0] = *(const U4*)(pr + hi * 16);
      pa.q[1] = *(const U4*)(pr + 8 + hi * 16);
    }

    // ---- O += P @ V, V B-fragments gathered from LDS (padded stride) ----
#pragma unroll
    for (int nt = 0; nt < 4; ++nt) {
      Frag bv;
#pragma unroll
      for (int j = 0; j < 16; ++j)
        bv.u[j] = Vlds[cur][(hi * 16 + j) * VP + nt * 16 + lo];
      o[nt] = __builtin_amdgcn_wmma_f32_16x16x32_bf16(
          false, pa.v, false, bv.v, (short)0, o[nt], false, false);
    }

#if HAVE_ASYNC_LDS
    asm volatile("s_wait_asynccnt 0x0" ::: "memory");   // next V tile landed
#endif
    __syncthreads();
  }

  // ---- finalize & store y (bf16, [B,T,C]) ----
#pragma unroll
  for (int v = 0; v < 8; ++v) {
    float inv = 1.0f / lrow[v];
    int t = qt * 16 + hi * 8 + v;
#pragma unroll
    for (int nt = 0; nt < 4; ++nt) {
      float val = o[nt][v] * inv;
      y[((size_t)(b * T + t)) * C + h * 64 + nt * 16 + lo] = f2bf(val);
    }
  }
}

// ---------- host launch ----------
extern "C" void kernel_launch(void* const* d_in, const int* in_sizes, int n_in,
                              void* d_out, int out_size, void* d_ws, size_t ws_size,
                              hipStream_t stream) {
  const int Bb = 4, T = 2048, C = 1024, H = 16;
  const int M = Bb * T;          // 8192
  const int C3 = 3 * C;          // 3072
  const int C4 = 4 * C;          // 4096

  const float* x     = (const float*)d_in[0];
  const float* ln1_g = (const float*)d_in[1];
  const float* ln1_b = (const float*)d_in[2];
  const float* W_qkv = (const float*)d_in[3];
  const float* b_qkv = (const float*)d_in[4];
  const float* W_o   = (const float*)d_in[5];
  const float* b_o   = (const float*)d_in[6];
  const float* ln2_g = (const float*)d_in[7];
  const float* ln2_b = (const float*)d_in[8];
  const float* W_fc  = (const float*)d_in[9];
  const float* b_fc  = (const float*)d_in[10];
  const float* W_pr  = (const float*)d_in[11];
  const float* b_pr  = (const float*)d_in[12];
  float* out = (float*)d_out;

  char* ws = (char*)d_ws;
  size_t off = 0;
  auto alloc = [&](size_t bytes) -> char* {
    char* p = ws + off;
    off += (bytes + 255) & ~(size_t)255;
    return p;
  };
  unsigned short* Wqkv_t = (unsigned short*)alloc((size_t)C  * C3 * 2);
  unsigned short* Wo_t   = (unsigned short*)alloc((size_t)C  * C  * 2);
  unsigned short* Wfc_t  = (unsigned short*)alloc((size_t)C  * C4 * 2);
  unsigned short* Wpr_t  = (unsigned short*)alloc((size_t)C4 * C  * 2);
  unsigned short* h1     = (unsigned short*)alloc((size_t)M * C  * 2);
  unsigned short* qkvb   = (unsigned short*)alloc((size_t)M * C3 * 2);
  unsigned short* yb     = (unsigned short*)alloc((size_t)M * C  * 2);
  float*          x1     = (float*)         alloc((size_t)M * C  * 4);
  unsigned short* h2     = (unsigned short*)alloc((size_t)M * C  * 2);
  unsigned short* fb     = (unsigned short*)alloc((size_t)M * C4 * 2);

  dim3 tb16(16, 16);
  // weight conversions (fp32 [K,N] -> bf16 [N,K])
  wconv_kernel<<<dim3(C3 / 16, C  / 16), tb16, 0, stream>>>(W_qkv, Wqkv_t, C,  C3);
  wconv_kernel<<<dim3(C  / 16, C  / 16), tb16, 0, stream>>>(W_o,   Wo_t,   C,  C);
  wconv_kernel<<<dim3(C4 / 16, C  / 16), tb16, 0, stream>>>(W_fc,  Wfc_t,  C,  C4);
  wconv_kernel<<<dim3(C  / 16, C4 / 16), tb16, 0, stream>>>(W_pr,  Wpr_t,  C4, C);

  // LN1
  ln_kernel<<<dim3(M), 256, 0, stream>>>(x, ln1_g, ln1_b, h1, C);

  // QKV projection: [M,C] @ [C,3C] -> bf16 [M,3C]
  gemm_bf16_kernel<1024, 0><<<dim3(C3 / 64, M / 64), 128, 0, stream>>>(
      h1, Wqkv_t, b_qkv, qkvb, nullptr, nullptr, C3);

  // flash attention -> yb bf16 [M,C]
  attn_kernel<<<dim3(T / 16, H, Bb), 32, 0, stream>>>(qkvb, yb, T, C3, C);

  // output projection + residual: x1 = x + yb @ W_o + b_o  (fp32)
  gemm_bf16_kernel<1024, 2><<<dim3(C / 64, M / 64), 128, 0, stream>>>(
      yb, Wo_t, b_o, nullptr, x1, x, C);

  // LN2
  ln_kernel<<<dim3(M), 256, 0, stream>>>(x1, ln2_g, ln2_b, h2, C);

  // FC + exact GELU -> fb bf16 [M,4C]
  gemm_bf16_kernel<1024, 1><<<dim3(C4 / 64, M / 64), 128, 0, stream>>>(
      h2, Wfc_t, b_fc, fb, nullptr, nullptr, C4);

  // projection + residual: out = x1 + fb @ W_pr + b_pr  (fp32)
  gemm_bf16_kernel<4096, 2><<<dim3(C / 64, M / 64), 128, 0, stream>>>(
      fb, Wpr_t, b_pr, nullptr, out, x1, C);
}
